// GEFA_30872224924331
// MI455X (gfx1250) — matrix operand order repaired and checked by reading, not compiled, not run on hardware
//
#include <hip/hip_runtime.h>
#include <hip/hip_bf16.h>

typedef unsigned short u16;
typedef __attribute__((ext_vector_type(16))) __bf16 v16bf;
typedef __attribute__((ext_vector_type(8)))  float  v8f;
typedef unsigned int v4u __attribute__((ext_vector_type(4)));
typedef int v8i __attribute__((ext_vector_type(8)));
typedef int v4i __attribute__((ext_vector_type(4)));

// Problem constants (from reference setup_inputs)
#define BB 256
#define LL 512
#define AA 40
#define PP 513          // L+1
#define ND 10240        // B*A
#define N2 131328       // B*P
#define EPB 2046        // edges per batch in protein graph: 2*(L-1)+2*L

// LDS tile: 64 rows x 32 bf16, row stride 40 halves (80B) via TDM padding
#define TILE_HALVES (64 * 40)
#define TILE_BYTES  (TILE_HALVES * 2)

__device__ __forceinline__ float bf2f(u16 h) {
  return __builtin_bit_cast(float, (unsigned)h << 16);
}
__device__ __forceinline__ u16 f2bf(float f) {
  unsigned u = __builtin_bit_cast(unsigned, f);
  unsigned r = u + 0x7FFFu + ((u >> 16) & 1u);
  return (u16)(r >> 16);
}

// ---------------------------------------------------------------------------
// Weight pack: f32 (K x M) -> bf16 B-fragment layout, zero padded to Kp x Mp.
// Frag (kb, nb) covers K rows kb*32..+31, cols nb*16..+15.
// Lane l supplies col n = nb*16 + (l&15); half h holds K = kb*32 + (l>=16)*16 + h.
// ---------------------------------------------------------------------------
__global__ void k_packw(const float* __restrict__ W, u16* __restrict__ out,
                        int K, int M, int Kp, int Mp) {
  const int frag = blockIdx.x;
  const int lane = threadIdx.x;
  const int NB = Mp >> 4;
  const int kb = frag / NB;
  const int nb = frag - kb * NB;
  const int n = nb * 16 + (lane & 15);
  const int kbase = kb * 32 + ((lane >> 4) << 4);
  u16* o = out + ((size_t)frag * 32 + lane) * 16;
#pragma unroll
  for (int h = 0; h < 16; ++h) {
    const int k = kbase + h;
    float v = (k < K && n < M) ? W[(size_t)k * M + n] : 0.0f;
    o[h] = f2bf(v);
  }
}

// ---------------------------------------------------------------------------
// GEMM: Out(N x Mp) = act(A(N x Kp, bf16) @ Wpacked + bias [+ Res]), bf16/f32 out.
// Block: 64 rows x (16*NT) cols, 128 threads (4 waves); wave w owns rows w*16..+15.
// A tiles DMA'd into double-buffered LDS by the Tensor Data Mover (wave 0 issues,
// TENSORcnt + workgroup barrier synchronize). TDM pad fields produce the 80B
// row stride directly. B frags load straight from packed weights (L2-resident).
// ---------------------------------------------------------------------------
template <int NT, int ACT, int BIAS, int RES, int OUTF32>
__global__ __launch_bounds__(128) void k_gemm(
    const u16* __restrict__ A, const u16* __restrict__ Wp,
    const float* __restrict__ bias, const u16* __restrict__ Res,
    void* __restrict__ Out, int Kp, int M, int Mp, int ldo) {
  extern __shared__ u16 Asd[];  // 2 * TILE_HALVES (dynamic, offset 0)

  const int n0 = blockIdx.x * (16 * NT);
  const int m0 = blockIdx.y * 64;
  const int lane = threadIdx.x & 31;
  const int wave = threadIdx.x >> 5;
  const int NB16 = Mp >> 4;
  const int nb0 = blockIdx.x * NT;

  v8f acc[NT];
#pragma unroll
  for (int t = 0; t < NT; ++t)
#pragma unroll
    for (int i = 0; i < 8; ++i) acc[t][i] = 0.0f;

  const int KB = Kp >> 5;

  // TDM descriptor issue: load 64x32 bf16 tile (k-block kt) into LDS buffer buf.
  // Row pad: after every 16 DWORDs (64B = one 32-elem row) insert 4 DWORDs ->
  // LDS row stride 80B, matching the fragment addressing below.
  auto tdm_issue = [&](int buf, int kt) {
    if (wave != 0) return;
    const unsigned long long ga =
        (unsigned long long)(const void*)(A + (size_t)m0 * Kp + (size_t)kt * 32);
    const v4u g0 = {
        1u,                                                  // count=1
        (unsigned)(buf * TILE_BYTES),                        // lds_addr
        (unsigned)(ga & 0xffffffffull),                      // global_addr lo
        (unsigned)((ga >> 32) & 0x01ffffffull) | (2u << 30)  // addr hi | type=2
    };
    const v8i g1 = {
        (int)((1u << 16) | (1u << 20) | (3u << 22) | (3u << 25)),  // 2B, pad 16dw+4dw
        (int)(((unsigned)Kp & 0xffffu) << 16),                     // tensor_dim0 lo
        (int)(((unsigned)Kp >> 16) & 0xffffu),                     // dim0 hi | dim1 lo
        (int)(16u | (32u << 16)),                                  // dim1 hi | tile_dim0=32
        (int)64,                                                   // tile_dim1=64
        (int)Kp,                                                   // dim0_stride lo
        0, 0};
    const v4i z4 = {0, 0, 0, 0};
    const v8i z8 = {0, 0, 0, 0, 0, 0, 0, 0};
    __builtin_amdgcn_tensor_load_to_lds(g0, g1, z4, z4, z8, 0);
  };

  tdm_issue(0, 0);
  if (wave == 0) __builtin_amdgcn_s_wait_tensorcnt(0);
  __syncthreads();

  // Per-lane fragment addresses (uniform strides, hoisted out of the loop)
  const int mr = wave * 16 + (lane & 15);
  const int k8 = (lane >> 4) << 4;
  const u16* wlane = Wp + ((size_t)nb0 * 32 + lane) * 16;  // frag (0, nb0)
  const size_t wstep = (size_t)NB16 * 512;                 // halves per k-block

  for (int kb = 0; kb < KB; ++kb) {
    const bool more = (kb + 1 < KB);
    if (more) tdm_issue((kb + 1) & 1, kb + 1);  // DMA next tile during compute

    // A fragment (16x32 bf16) from current LDS buffer
    union { uint2 u[4]; v16bf v; } af;
    {
      const u16* asb = Asd + (kb & 1) * TILE_HALVES;
      const uint2* p0 = (const uint2*)(&asb[mr * 40 + k8]);
      const uint2* p1 = (const uint2*)(&asb[mr * 40 + 8 + k8]);
      af.u[0] = p0[0];
      af.u[1] = p0[1];
      af.u[2] = p1[0];
      af.u[3] = p1[1];
    }

    // B fragments: one clause of coalesced 32B loads, then NT WMMAs
    union { uint4 u[2]; v16bf v; } bf[NT];
#pragma unroll
    for (int t = 0; t < NT; ++t) {
      const uint4* wp4 =
          (const uint4*)(wlane + (size_t)kb * wstep + (size_t)t * 512);
      bf[t].u[0] = wp4[0];
      bf[t].u[1] = wp4[1];
    }
#pragma unroll
    for (int t = 0; t < NT; ++t)
      acc[t] = __builtin_amdgcn_wmma_f32_16x16x32_bf16(
          false, af.v, false, bf[t].v, (short)0, acc[t], false, false);

    if (wave == 0 && more) __builtin_amdgcn_s_wait_tensorcnt(0);
    __syncthreads();
  }

  // Epilogue: C layout lane l -> col (l&15); VGPR v -> row v + (l>=16)*8
  const int rbase = m0 + wave * 16 + ((lane >> 4) << 3);
#pragma unroll
  for (int t = 0; t < NT; ++t) {
    const int col = n0 + t * 16 + (lane & 15);
    const float bv = (BIAS && col < M) ? bias[col] : 0.0f;
#pragma unroll
    for (int v = 0; v < 8; ++v) {
      const int row = rbase + v;
      float x = acc[t][v] + bv;
      if (RES) x += bf2f(Res[(size_t)row * ldo + col]);
      if (ACT == 1) x = fmaxf(x, 0.0f);
      if (ACT == 2) x = tanhf(x);
      if (col >= M) x = 0.0f;
      if (OUTF32) {
        if (col < M) ((float*)Out)[(size_t)row * ldo + col] = x;
      } else {
        ((u16*)Out)[(size_t)row * ldo + col] = f2bf(x);
      }
    }
  }
}

// Host-side NT dispatch: pick the widest wave tile that divides Mp/16.
template <int ACT, int BIAS, int RES, int OUTF32>
static void launch_gemm(hipStream_t stream, const u16* A, const u16* Wp,
                        const float* bias, const u16* Res, void* Out, int N,
                        int Kp, int M, int Mp, int ldo) {
  const int NB16 = Mp >> 4;
  dim3 blk(128);
  const size_t shm = 2 * TILE_BYTES;
  if ((NB16 & 3) == 0) {
    dim3 g((unsigned)(NB16 / 4), (unsigned)(N / 64));
    k_gemm<4, ACT, BIAS, RES, OUTF32><<<g, blk, shm, stream>>>(A, Wp, bias, Res, Out, Kp, M, Mp, ldo);
  } else if ((NB16 & 1) == 0) {
    dim3 g((unsigned)(NB16 / 2), (unsigned)(N / 64));
    k_gemm<2, ACT, BIAS, RES, OUTF32><<<g, blk, shm, stream>>>(A, Wp, bias, Res, Out, Kp, M, Mp, ldo);
  } else {
    dim3 g((unsigned)NB16, (unsigned)(N / 64));
    k_gemm<1, ACT, BIAS, RES, OUTF32><<<g, blk, shm, stream>>>(A, Wp, bias, Res, Out, Kp, M, Mp, ldo);
  }
}

// ---------------------------------------------------------------------------
// Input builders
// ---------------------------------------------------------------------------
__global__ void k_build_drug(const float* __restrict__ x, u16* __restrict__ xb) {
  const int node = blockIdx.x, col = threadIdx.x;  // <<<ND, 96>>>
  float v = (col < 78) ? x[(size_t)node * 78 + col] : 0.0f;
  xb[(size_t)node * 96 + col] = f2bf(v);
}
__global__ void k_build_prot(const float* __restrict__ x, u16* __restrict__ xb) {
  const int node = blockIdx.x, col = threadIdx.x;  // <<<N2, 64>>>
  float v = (col < 54) ? x[(size_t)node * 54 + col] : 0.0f;
  xb[(size_t)node * 64 + col] = f2bf(v);
}
__global__ void k_set_drug_rows(const u16* __restrict__ dvec, u16* __restrict__ xb) {
  const int b = blockIdx.x, col = threadIdx.x;  // <<<BB, 64>>>
  xb[((size_t)b * PP + LL) * 64 + col] = dvec[(size_t)b * 64 + col];
}

// ---------------------------------------------------------------------------
// Drug GCN aggregate (chain of 40, unit weights): out = relu(agg + h*dinv^2 + b)
// ---------------------------------------------------------------------------
__device__ __forceinline__ float ddinv(int a) {
  return rsqrtf(1.0f + (a > 0 ? 1.0f : 0.0f) + (a < AA - 1 ? 1.0f : 0.0f));
}
__global__ void k_drug_agg(const u16* __restrict__ hb, u16* __restrict__ out,
                           const float* __restrict__ bias, int S, int M) {
  const int node = blockIdx.x, col = threadIdx.x;  // <<<ND, S>>>
  const int a = node % AA;
  const float ds = ddinv(a);
  float acc = 0.0f;
  if (a > 0)      acc += bf2f(hb[(size_t)(node - 1) * S + col]) * ddinv(a - 1);
  if (a < AA - 1) acc += bf2f(hb[(size_t)(node + 1) * S + col]) * ddinv(a + 1);
  float x = ds * acc + bf2f(hb[(size_t)node * S + col]) * ds * ds +
            (col < M ? bias[col] : 0.0f);
  out[(size_t)node * S + col] = (col < M) ? f2bf(fmaxf(x, 0.0f)) : (u16)0;
}

// ---------------------------------------------------------------------------
// Protein dinv (weighted: prot_ea + attention; unweighted: ones)
// ---------------------------------------------------------------------------
__global__ void k_prot_dinv(const float* __restrict__ ea,
                            const float* __restrict__ att,
                            float* __restrict__ dinv, int weighted) {
  const int idx = blockIdx.x * blockDim.x + threadIdx.x;
  if (idx >= N2) return;
  const int b = idx / PP, j = idx - b * PP;
  float deg;
  if (j == LL) {
    deg = weighted ? 2.0f : (float)(LL + 1);  // softmax sums to 1
  } else if (weighted) {
    const float* e = ea + (size_t)b * EPB;
    deg = 1.0f + (j > 0 ? e[j - 1] : 0.0f) + (j < LL - 1 ? e[(LL - 1) + j] : 0.0f) +
          att[b * LL + j];
  } else {
    deg = 2.0f + (j > 0 ? 1.0f : 0.0f) + (j < LL - 1 ? 1.0f : 0.0f);
  }
  dinv[idx] = rsqrtf(deg);
}

// Residue nodes: gather from j-1, j+1 (chain) and hub drug node.
__global__ void k_prot_res_agg(const u16* __restrict__ hb, u16* __restrict__ out,
                               const float* __restrict__ bias,
                               const float* __restrict__ ea,
                               const float* __restrict__ att,
                               const float* __restrict__ dinv, int S, int M) {
  const int node = blockIdx.x, col = threadIdx.x;  // <<<BB*LL, S>>>
  const int b = node >> 9, j = node & (LL - 1);
  const size_t row = (size_t)b * PP + j;
  const float ds = dinv[row];
  float acc = 0.0f;
  if (j > 0) {
    float w = ea ? ea[(size_t)b * EPB + (j - 1)] : 1.0f;
    acc += w * bf2f(hb[(row - 1) * S + col]) * dinv[row - 1];
  }
  if (j < LL - 1) {
    float w = ea ? ea[(size_t)b * EPB + (LL - 1) + j] : 1.0f;
    acc += w * bf2f(hb[(row + 1) * S + col]) * dinv[row + 1];
  }
  {
    float w = att ? att[b * LL + j] : 1.0f;
    const size_t dr = (size_t)b * PP + LL;
    acc += w * bf2f(hb[dr * S + col]) * dinv[dr];
  }
  float x = ds * acc + bf2f(hb[row * S + col]) * ds * ds +
            (col < M ? bias[col] : 0.0f);
  out[row * S + col] = (col < M) ? f2bf(fmaxf(x, 0.0f)) : (u16)0;
}

// Hub drug node: reduce over all 512 residues of its batch.
__global__ void k_prot_drug_agg(const u16* __restrict__ hb, u16* __restrict__ out,
                                const float* __restrict__ bias,
                                const float* __restrict__ att,
                                const float* __restrict__ dinv, int S, int M) {
  const int b = blockIdx.x, col = threadIdx.x;  // <<<BB, S>>>
  const size_t dr = (size_t)b * PP + LL;
  const float ds = dinv[dr];
  float acc = 0.0f;
  for (int j = 0; j < LL; ++j) {
    const float w = att ? att[b * LL + j] : 1.0f;
    const size_t r = (size_t)b * PP + j;
    acc += w * bf2f(hb[r * S + col]) * dinv[r];
  }
  float x = ds * acc + bf2f(hb[dr * S + col]) * ds * ds +
            (col < M ? bias[col] : 0.0f);
  out[dr * S + col] = (col < M) ? f2bf(fmaxf(x, 0.0f)) : (u16)0;
}

// ---------------------------------------------------------------------------
// Attention: logits = tanh_out(54) . W_att2 + b; softmax over 512 residues
// ---------------------------------------------------------------------------
__global__ void k_att_logits(const u16* __restrict__ tb, const float* __restrict__ w2,
                             const float* __restrict__ b2, float* __restrict__ lo) {
  const int idx = blockIdx.x * blockDim.x + threadIdx.x;  // over BB*LL
  if (idx >= BB * LL) return;
  const int b = idx >> 9, j = idx & (LL - 1);
  const u16* p = tb + ((size_t)b * PP + j) * 64;
  float s = b2[0];
  for (int k = 0; k < 54; ++k) s += bf2f(p[k]) * w2[k];
  lo[idx] = s;
}
__global__ void k_softmax512(const float* __restrict__ lo, float* __restrict__ att) {
  __shared__ float red[256];
  const int b = blockIdx.x, t = threadIdx.x;
  const float a0 = lo[b * LL + t], a1 = lo[b * LL + 256 + t];
  red[t] = fmaxf(a0, a1);
  __syncthreads();
  for (int s = 128; s > 0; s >>= 1) {
    if (t < s) red[t] = fmaxf(red[t], red[t + s]);
    __syncthreads();
  }
  const float mx = red[0];
  __syncthreads();
  const float e0 = __expf(a0 - mx), e1 = __expf(a1 - mx);
  red[t] = e0 + e1;
  __syncthreads();
  for (int s = 128; s > 0; s >>= 1) {
    if (t < s) red[t] += red[t + s];
    __syncthreads();
  }
  const float inv = 1.0f / red[0];
  att[b * LL + t] = e0 * inv;
  att[b * LL + 256 + t] = e1 * inv;
}

// ---------------------------------------------------------------------------
// Pooling / head glue
// ---------------------------------------------------------------------------
__global__ void k_drug_pool(const u16* __restrict__ xb, u16* __restrict__ xg) {
  const int b = blockIdx.x, col = threadIdx.x;  // <<<BB, 160>>>
  float m = -1e30f;
  for (int a = 0; a < AA; ++a)
    m = fmaxf(m, bf2f(xb[((size_t)b * AA + a) * 160 + col]));
  xg[(size_t)b * 160 + col] = f2bf(m);
}
__global__ void k_prot_pool(const u16* __restrict__ xb, u16* __restrict__ x2g,
                            u16* __restrict__ dafter) {
  const int b = blockIdx.x, col = threadIdx.x;  // <<<BB, 256>>>
  float m = -1e30f;
  for (int j = 0; j < LL; ++j)
    m = fmaxf(m, bf2f(xb[((size_t)b * PP + j) * 256 + col]));
  x2g[(size_t)b * 256 + col] = f2bf(m);
  dafter[(size_t)b * 256 + col] = xb[((size_t)b * PP + LL) * 256 + col];
}
__global__ void k_concat(const u16* __restrict__ dafter, const u16* __restrict__ xchg,
                         const u16* __restrict__ x2v, u16* __restrict__ xc) {
  const int b = blockIdx.x, c = threadIdx.x;  // <<<BB, 512>>>
  u16 v;
  if (c < 256)
    v = f2bf(fmaxf(bf2f(dafter[(size_t)b * 256 + c]), bf2f(xchg[(size_t)b * 256 + c])));
  else
    v = x2v[(size_t)b * 256 + (c - 256)];
  xc[(size_t)b * 512 + c] = v;
}

// ---------------------------------------------------------------------------
// Host driver
// ---------------------------------------------------------------------------
extern "C" void kernel_launch(void* const* d_in, const int* in_sizes, int n_in,
                              void* d_out, int out_size, void* d_ws, size_t ws_size,
                              hipStream_t stream) {
  (void)in_sizes; (void)n_in; (void)out_size; (void)ws_size;
  char* ws = (char*)d_ws;
  size_t cur = 0;
  auto alloc = [&](size_t b) { size_t o = cur; cur += (b + 255) & ~(size_t)255; return o; };

  const float* drug_x = (const float*)d_in[0];
  const float* prot_x = (const float*)d_in[3];
  const float* prot_ea = (const float*)d_in[6];

  auto packw = [&](int idx, int K, int M, int Kp, int Mp) -> const u16* {
    u16* dst = (u16*)(ws + alloc((size_t)Kp * Mp * 2));
    const int frags = (Kp / 32) * (Mp / 16);
    k_packw<<<dim3(frags), dim3(32), 0, stream>>>((const float*)d_in[idx], dst, K, M, Kp, Mp);
    return dst;
  };

  const u16* Wc1d  = packw(7,  78,   78,   96,   96);
  const u16* Wc2d  = packw(9,  78,   156,  96,   160);
  const u16* Wrdg  = packw(11, 156,  156,  160,  160);
  const u16* Wrdl  = packw(13, 156,  156,  160,  160);
  const u16* Wfg1d = packw(14, 156,  1024, 160,  1024);
  const u16* Wfg2d = packw(16, 1024, 54,   1024, 64);
  const u16* Wfg3d = packw(18, 54,   256,  64,   256);
  const u16* Watt1 = packw(20, 54,   54,   64,   64);
  const u16* Wc1t  = packw(24, 54,   128,  64,   128);
  const u16* Wc2t  = packw(26, 128,  256,  128,  256);
  const u16* Wrtg  = packw(28, 256,  256,  256,  256);
  const u16* Wrtl  = packw(30, 256,  256,  256,  256);
  const u16* Wfg1t = packw(31, 256,  1024, 256,  1024);
  const u16* Wfg2t = packw(33, 1024, 256,  1024, 256);
  const u16* Wfc1  = packw(35, 512,  1024, 512,  1024);
  const u16* Wfc2  = packw(37, 1024, 512,  1024, 512);
  const u16* Wout  = packw(39, 512,  1,    512,  16);

  const float* b_c1d  = (const float*)d_in[8];
  const float* b_c2d  = (const float*)d_in[10];
  const float* b_rd_g = (const float*)d_in[12];
  const float* b_fg1d = (const float*)d_in[15];
  const float* b_fg2d = (const float*)d_in[17];
  const float* b_fg3d = (const float*)d_in[19];
  const float* b_att1 = (const float*)d_in[21];
  const float* w_att2 = (const float*)d_in[22];
  const float* b_att2 = (const float*)d_in[23];
  const float* b_c1t  = (const float*)d_in[25];
  const float* b_c2t  = (const float*)d_in[27];
  const float* b_rt_g = (const float*)d_in[29];
  const float* b_fg1t = (const float*)d_in[32];
  const float* b_fg2t = (const float*)d_in[34];
  const float* b_fc1  = (const float*)d_in[36];
  const float* b_fc2  = (const float*)d_in[38];
  const float* b_out  = (const float*)d_in[40];

  u16* xb_d = (u16*)(ws + alloc((size_t)ND * 160 * 2));
  u16* hb_d = (u16*)(ws + alloc((size_t)ND * 160 * 2));
  u16* tb_d = (u16*)(ws + alloc((size_t)ND * 160 * 2));
  u16* xb_p = (u16*)(ws + alloc((size_t)N2 * 256 * 2));
  u16* hb_p = (u16*)(ws + alloc((size_t)N2 * 256 * 2));
  u16* tb_p = (u16*)(ws + alloc((size_t)N2 * 256 * 2));
  float* dinv   = (float*)(ws + alloc((size_t)N2 * 4));
  float* logits = (float*)(ws + alloc((size_t)BB * LL * 4));
  float* att    = (float*)(ws + alloc((size_t)BB * LL * 4));
  u16* xg     = (u16*)(ws + alloc((size_t)BB * 160 * 2));
  u16* t1024  = (u16*)(ws + alloc((size_t)BB * 1024 * 2));
  u16* fg2o   = (u16*)(ws + alloc((size_t)BB * 64 * 2));
  u16* xchg   = (u16*)(ws + alloc((size_t)BB * 256 * 2));
  u16* x2g    = (u16*)(ws + alloc((size_t)BB * 256 * 2));
  u16* dafter = (u16*)(ws + alloc((size_t)BB * 256 * 2));
  u16* x2v    = (u16*)(ws + alloc((size_t)BB * 256 * 2));
  u16* xc     = (u16*)(ws + alloc((size_t)BB * 512 * 2));
  u16* fc2o   = (u16*)(ws + alloc((size_t)BB * 512 * 2));

  auto gemm = [&](const u16* A, const u16* Wp, const float* bias, const u16* Res,
                  void* Out, int N, int Kp, int M, int Mp, int ldo, int act,
                  int outf32) {
    if (outf32)
      launch_gemm<0, 1, 0, 1>(stream, A, Wp, bias, Res, Out, N, Kp, M, Mp, ldo);
    else if (Res)
      launch_gemm<1, 0, 1, 0>(stream, A, Wp, bias, Res, Out, N, Kp, M, Mp, ldo);
    else if (act == 2)
      launch_gemm<2, 1, 0, 0>(stream, A, Wp, bias, Res, Out, N, Kp, M, Mp, ldo);
    else if (act == 1)
      launch_gemm<1, 1, 0, 0>(stream, A, Wp, bias, Res, Out, N, Kp, M, Mp, ldo);
    else if (bias)
      launch_gemm<0, 1, 0, 0>(stream, A, Wp, bias, Res, Out, N, Kp, M, Mp, ldo);
    else
      launch_gemm<0, 0, 0, 0>(stream, A, Wp, bias, Res, Out, N, Kp, M, Mp, ldo);
  };

  // ---------------- Drug branch ----------------
  k_build_drug<<<ND, 96, 0, stream>>>(drug_x, xb_d);
  gemm(xb_d, Wc1d, nullptr, nullptr, hb_d, ND, 96, 78, 96, 96, 0, 0);
  k_drug_agg<<<ND, 96, 0, stream>>>(hb_d, tb_d, b_c1d, 96, 78);
  gemm(tb_d, Wc2d, nullptr, nullptr, hb_d, ND, 96, 156, 160, 160, 0, 0);
  k_drug_agg<<<ND, 160, 0, stream>>>(hb_d, xb_d, b_c2d, 160, 156);
  for (int r = 0; r < 4; ++r) {
    gemm(xb_d, Wrdg, nullptr, nullptr, hb_d, ND, 160, 156, 160, 160, 0, 0);
    k_drug_agg<<<ND, 160, 0, stream>>>(hb_d, tb_d, b_rd_g, 160, 156);
    gemm(tb_d, Wrdl, nullptr, xb_d, xb_d, ND, 160, 156, 160, 160, 1, 0);
  }
  k_drug_pool<<<BB, 160, 0, stream>>>(xb_d, xg);
  gemm(xg, Wfg1d, b_fg1d, nullptr, t1024, BB, 160, 1024, 1024, 1024, 1, 0);
  gemm(t1024, Wfg2d, b_fg2d, nullptr, fg2o, BB, 1024, 54, 64, 64, 0, 0);
  gemm(fg2o, Wfg3d, b_fg3d, nullptr, xchg, BB, 64, 256, 256, 256, 1, 0);

  // ---------------- Attention ----------------
  k_build_prot<<<N2, 64, 0, stream>>>(prot_x, xb_p);
  gemm(xb_p, Watt1, b_att1, nullptr, hb_p, N2, 64, 54, 64, 64, 2, 0);
  k_att_logits<<<(BB * LL + 255) / 256, 256, 0, stream>>>(hb_p, w_att2, b_att2, logits);
  k_softmax512<<<BB, 256, 0, stream>>>(logits, att);

  // ---------------- Protein branch ----------------
  k_set_drug_rows<<<BB, 64, 0, stream>>>(fg2o, xb_p);
  k_prot_dinv<<<(N2 + 255) / 256, 256, 0, stream>>>(prot_ea, att, dinv, 1);
  gemm(xb_p, Wc1t, nullptr, nullptr, hb_p, N2, 64, 128, 128, 128, 0, 0);
  k_prot_res_agg<<<BB * LL, 128, 0, stream>>>(hb_p, tb_p, b_c1t, prot_ea, att, dinv, 128, 128);
  k_prot_drug_agg<<<BB, 128, 0, stream>>>(hb_p, tb_p, b_c1t, att, dinv, 128, 128);
  gemm(tb_p, Wc2t, nullptr, nullptr, hb_p, N2, 128, 256, 256, 256, 0, 0);
  k_prot_res_agg<<<BB * LL, 256, 0, stream>>>(hb_p, xb_p, b_c2t, prot_ea, att, dinv, 256, 256);
  k_prot_drug_agg<<<BB, 256, 0, stream>>>(hb_p, xb_p, b_c2t, att, dinv, 256, 256);
  k_prot_dinv<<<(N2 + 255) / 256, 256, 0, stream>>>(nullptr, nullptr, dinv, 0);
  for (int r = 0; r < 4; ++r) {
    gemm(xb_p, Wrtg, nullptr, nullptr, hb_p, N2, 256, 256, 256, 256, 0, 0);
    k_prot_res_agg<<<BB * LL, 256, 0, stream>>>(hb_p, tb_p, b_rt_g, nullptr, nullptr, dinv, 256, 256);
    k_prot_drug_agg<<<BB, 256, 0, stream>>>(hb_p, tb_p, b_rt_g, nullptr, dinv, 256, 256);
    gemm(tb_p, Wrtl, nullptr, xb_p, xb_p, N2, 256, 256, 256, 256, 1, 0);
  }

  // ---------------- Head ----------------
  k_prot_pool<<<BB, 256, 0, stream>>>(xb_p, x2g, dafter);
  gemm(x2g, Wfg1t, b_fg1t, nullptr, t1024, BB, 256, 1024, 1024, 1024, 1, 0);
  gemm(t1024, Wfg2t, b_fg2t, nullptr, x2v, BB, 1024, 256, 256, 256, 0, 0);
  k_concat<<<BB, 512, 0, stream>>>(dafter, xchg, x2v, xc);
  gemm(xc, Wfc1, b_fc1, nullptr, t1024, BB, 512, 1024, 1024, 1024, 1, 0);
  gemm(t1024, Wfc2, b_fc2, nullptr, fc2o, BB, 1024, 512, 512, 512, 1, 0);
  gemm(fc2o, Wout, b_out, nullptr, d_out, BB, 512, 1, 16, 1, 0, 1);
}
